// SimpleModel_34634616275158
// MI455X (gfx1250) — matrix operand order repaired and checked by reading, not compiled, not run on hardware
//
#include <hip/hip_runtime.h>
#include <hip/hip_bf16.h>

typedef __attribute__((ext_vector_type(16))) _Float16 v16h;
typedef __attribute__((ext_vector_type(8)))  float    v8f;

// ---------- helpers ----------
__device__ __forceinline__ unsigned floatOrd(float f) {
    unsigned u = __float_as_uint(f);
    return (u & 0x80000000u) ? ~u : (u | 0x80000000u);
}
__device__ __forceinline__ float ordFloat(unsigned u) {
    return (u & 0x80000000u) ? __uint_as_float(u & 0x7fffffffu) : __uint_as_float(~u);
}
__device__ __forceinline__ float leaky02(float a) { return (a > 0.f) ? a : 0.2f * a; }

// CDNA5 16-bit A fragment (16x32, MxK): lane 0-15 -> M=lane, K in {0..7,16..23};
// lane 16-31 -> M=lane-16, K in {8..15,24..31}. Per-lane K ranges are contiguous
// in a row-major tile -> compiler coalesces to ds_load_b128.
__device__ __forceinline__ v16h lds_load_A(const _Float16* tile, int ldk, int k0, int lane) {
    int m = lane & 15, g = (lane >> 4) & 1;
    v16h a;
#pragma unroll
    for (int t = 0; t < 16; ++t) {
        int k = k0 + (t & 7) + 8 * g + ((t & 8) ? 16 : 0);
        a[t] = tile[m * ldk + k];
    }
    return a;
}
// B fragment from LDS pre-swizzled into fragment order: wf[lane*16 + t].
// 32 contiguous, 32B-aligned bytes per lane -> 2x ds_load_b128.
__device__ __forceinline__ v16h lds_load_B_swz(const _Float16* wf, int lane) {
    return *(const v16h*)(wf + lane * 16);
}
__device__ __forceinline__ v8f wmma_f16(v16h a, v16h b, v8f c) {
    return __builtin_amdgcn_wmma_f32_16x16x32_f16(false, a, false, b, (short)0, c, false, false);
}

// ---------- init / small precompute ----------
__global__ void init1_kernel(unsigned* menc, float* z, float* x1, float* x2,
                             const float* b1, const float* b2, float* scal, int N) {
    int i = blockIdx.x * blockDim.x + threadIdx.x;
    if (i >= N) return;
    menc[i] = 0u;
    z[i] = 0.f;
#pragma unroll
    for (int k = 0; k < 16; ++k) x1[(size_t)i * 16 + k] = b1[k];
    x2[(size_t)i * 2 + 0] = b2[0];
    x2[(size_t)i * 2 + 1] = b2[1];
    if (i < 16) scal[i] = 0.f;
}
__global__ void init2_kernel(unsigned* menc, float* z, int N) {
    int i = blockIdx.x * blockDim.x + threadIdx.x;
    if (i >= N) return;
    menc[i] = 0u;
    z[i] = 0.f;
}
// scal[7..9] = We1 @ ae1 ; scal[10..13] = We2 @ ae2
__global__ void precompute_kernel(const float* We1, const float* ae1,
                                  const float* We2, const float* ae2, float* scal) {
    int t = threadIdx.x;
    if (t < 3) {
        float s = 0.f;
        for (int k = 0; k < 16; ++k) s += We1[t * 16 + k] * ae1[k];
        scal[7 + t] = s;
    } else if (t < 7) {
        int j = t - 3;
        float s = 0.f;
        for (int k = 0; k < 2; ++k) s += We2[j * 2 + k] * ae2[k];
        scal[10 + j] = s;
    }
}
__global__ void mean_ea_kernel(const float* ea, float* scal, int E) {
    float s0 = 0.f, s1 = 0.f, s2 = 0.f;
    for (int i = blockIdx.x * blockDim.x + threadIdx.x; i < E; i += gridDim.x * blockDim.x) {
        s0 += ea[(size_t)i * 3 + 0];
        s1 += ea[(size_t)i * 3 + 1];
        s2 += ea[(size_t)i * 3 + 2];
    }
    atomicAdd(&scal[0], s0); atomicAdd(&scal[1], s1); atomicAdd(&scal[2], s2);
}
__global__ void mean_e_kernel(const float* er, float* scal, int E) {
    float s0 = 0.f, s1 = 0.f, s2 = 0.f, s3 = 0.f;
    for (int i = blockIdx.x * blockDim.x + threadIdx.x; i < E; i += gridDim.x * blockDim.x) {
        s0 += er[(size_t)i * 4 + 0];
        s1 += er[(size_t)i * 4 + 1];
        s2 += er[(size_t)i * 4 + 2];
        s3 += er[(size_t)i * 4 + 3];
    }
    atomicAdd(&scal[3], s0); atomicAdd(&scal[4], s1);
    atomicAdd(&scal[5], s2); atomicAdd(&scal[6], s3);
}

// ---------- GATConv 1 ----------
__global__ void node1_kernel(const float* x, const float* W1, const float* as1,
                             const float* ad1, float* h1, float* hs1, float* hd1, int N) {
    int i = blockIdx.x * blockDim.x + threadIdx.x;
    if (i >= N) return;
    float x0 = x[(size_t)i * 4 + 0], x1v = x[(size_t)i * 4 + 1];
    float x2v = x[(size_t)i * 4 + 2], x3 = x[(size_t)i * 4 + 3];
    float hs = 0.f, hd = 0.f;
#pragma unroll
    for (int k = 0; k < 16; ++k) {
        float h = x0 * W1[k] + x1v * W1[16 + k] + x2v * W1[32 + k] + x3 * W1[48 + k];
        h1[(size_t)i * 16 + k] = h;
        hs += h * as1[k];
        hd += h * ad1[k];
    }
    hs1[i] = hs; hd1[i] = hd;
}
__global__ void alpha1_kernel(const int* ei, const float* ea, const float* hs1,
                              const float* hd1, const float* scal, float* alpha,
                              unsigned* menc, int E, int N) {
    int idx = blockIdx.x * blockDim.x + threadIdx.x;
    if (idx >= E + N) return;
    float v0 = scal[7], v1 = scal[8], v2 = scal[9];
    int s, d; float et;
    if (idx < E) {
        s = ei[idx]; d = ei[(size_t)E + idx];
        const float* a_ = ea + (size_t)idx * 3;
        et = a_[0] * v0 + a_[1] * v1 + a_[2] * v2;
    } else {
        int i = idx - E; s = i; d = i;
        float invE = 1.f / (float)E;
        et = (scal[0] * v0 + scal[1] * v1 + scal[2] * v2) * invE;
    }
    float a = leaky02(hs1[s] + hd1[d] + et);
    alpha[idx] = a;
    atomicMax(&menc[d], floatOrd(a));
}
__global__ void exp_kernel(const int* ei, float* alpha, const unsigned* menc,
                           float* z, int E, int N) {
    int idx = blockIdx.x * blockDim.x + threadIdx.x;
    if (idx >= E + N) return;
    int d = (idx < E) ? ei[(size_t)E + idx] : (idx - E);
    float p = expf(alpha[idx] - ordFloat(menc[d]));
    alpha[idx] = p;
    atomicAdd(&z[d], p);
}
__global__ void agg1_kernel(const int* ei, const float* p, const float* z,
                            const float* h1, float* x1, int E, int N) {
    int idx = blockIdx.x * blockDim.x + threadIdx.x;
    if (idx >= E + N) return;
    int s, d;
    if (idx < E) { s = ei[idx]; d = ei[(size_t)E + idx]; }
    else { s = idx - E; d = s; }
    float w = p[idx] / z[d];
    const float* hs = h1 + (size_t)s * 16;
    float* xo = x1 + (size_t)d * 16;
#pragma unroll
    for (int k = 0; k < 16; ++k) atomicAdd(&xo[k], w * hs[k]);
}

// ---------- edge MLP (WMMA f16, f32 accumulate) ----------
__global__ void __launch_bounds__(256)
edge_mlp_kernel(const int* ei, const float* ea, const float* x1,
                const float* Wm1, const float* bm1, const float* Wm2, const float* bm2,
                const float* Wm3, const float* bm3, const float* Wm4, const float* bm4,
                float* e_raw, float* out_e, int E) {
    // Weights pre-swizzled into B-fragment order: wf[lane*16 + t] = W[k(t,lane)][n(lane)]
    __shared__ alignas(32) _Float16 w1f[2][512];  // layer1: K=0..31 and K=32..63 steps
    __shared__ alignas(32) _Float16 w2f[512];
    __shared__ alignas(32) _Float16 w3f[512];
    __shared__ alignas(32) _Float16 w4f[512];
    __shared__ float bs[44];                      // bm1[0..15] bm2[16..31] bm3[32..39] bm4[40..43]
    __shared__ alignas(32) _Float16 zbuf[8][16 * 64]; // per-wave zcat tile, K padded to 64
    __shared__ alignas(32) _Float16 ibuf[8][16 * 32]; // per-wave inter-layer act, K padded to 32
    __shared__ float cbuf[8][16 * 4];                 // per-wave final 16x4 result

    int tid = threadIdx.x;
    // stage swizzled weight fragments
    for (int t = tid; t < 512; t += 256) {
        int lane = t >> 4, idx = t & 15;
        int n = lane & 15, g = (lane >> 4) & 1;
        int k = (idx & 7) + 8 * g + ((idx & 8) ? 16 : 0); // K within one 32-step
        w1f[0][t] = (_Float16)(Wm1[k * 16 + n]);                       // k<32 always valid (<35)
        int k2 = 32 + k;
        w1f[1][t] = (_Float16)((k2 < 35) ? Wm1[k2 * 16 + n] : 0.f);
        w2f[t] = (_Float16)((k < 16) ? Wm2[k * 16 + n] : 0.f);
        w3f[t] = (_Float16)((k < 16 && n < 8) ? Wm3[k * 8 + n] : 0.f);
        w4f[t] = (_Float16)((k < 8 && n < 4) ? Wm4[k * 4 + n] : 0.f);
    }
    if (tid < 44) {
        float v;
        if (tid < 16) v = bm1[tid];
        else if (tid < 32) v = bm2[tid - 16];
        else if (tid < 40) v = bm3[tid - 32];
        else v = bm4[tid - 40];
        bs[tid] = v;
    }
    __syncthreads();

    int wav = tid >> 5, lane = tid & 31;
    _Float16* zb = zbuf[wav];
    _Float16* ib = ibuf[wav];
    float*    cb = cbuf[wav];
    long tile = (long)blockIdx.x * 8 + wav;
    long e0 = tile * 16;

    int j = lane & 15, g = (lane >> 4) & 1;
    int n = j;

    // zero the padded K region of ibuf (cols 16..31) once
#pragma unroll
    for (int r = 0; r < 8; ++r) ib[(r + 8 * g) * 32 + 16 + j] = (_Float16)0.f;

    // stage zcat = [x1[src](16) | edge_attr(3) | x1[dst](16) | 0-pad] as f16
    {
        long eidx = e0 + j;
        if (eidx >= E) eidx = (long)E - 1;
        int si = ei[eidx], di = ei[(long)E + eidx];
        const float* xs = x1 + (size_t)si * 16;
        const float* xd = x1 + (size_t)di * 16;
        const float* ez = ea + (size_t)eidx * 3;
        for (int kk = 0; kk < 32; ++kk) {
            int k = g * 32 + kk;
            float v;
            if (k < 16) v = xs[k];
            else if (k < 19) v = ez[k - 16];
            else if (k < 35) v = xd[k - 19];
            else v = 0.f;
            zb[j * 64 + k] = (_Float16)v;
        }
    }
    asm volatile("s_wait_dscnt 0x0" ::: "memory");

    v8f zero = {};
    // layer 1: [16x35] @ [35x16] (K padded to 64 -> 2 WMMA)
    v8f c = zero;
    c = wmma_f16(lds_load_A(zb, 64, 0, lane),  lds_load_B_swz(w1f[0], lane), c);
    c = wmma_f16(lds_load_A(zb, 64, 32, lane), lds_load_B_swz(w1f[1], lane), c);
    {
        float bv = bs[n];
#pragma unroll
        for (int r = 0; r < 8; ++r) {
            float v = fmaxf(c[r] + bv, 0.f);
            ib[(r + 8 * g) * 32 + n] = (_Float16)v;
        }
    }
    asm volatile("s_wait_dscnt 0x0" ::: "memory");
    // layer 2: [16x16] @ [16x16]
    c = zero;
    c = wmma_f16(lds_load_A(ib, 32, 0, lane), lds_load_B_swz(w2f, lane), c);
    {
        float bv = bs[16 + n];
#pragma unroll
        for (int r = 0; r < 8; ++r) {
            float v = fmaxf(c[r] + bv, 0.f);
            ib[(r + 8 * g) * 32 + n] = (_Float16)v;
        }
    }
    asm volatile("s_wait_dscnt 0x0" ::: "memory");
    // layer 3: [16x16] @ [16x8] (N padded to 16)
    c = zero;
    c = wmma_f16(lds_load_A(ib, 32, 0, lane), lds_load_B_swz(w3f, lane), c);
    {
        float bv = (n < 8) ? bs[32 + n] : 0.f;
#pragma unroll
        for (int r = 0; r < 8; ++r) {
            float v = fmaxf(c[r] + bv, 0.f);   // cols >=8 become 0
            ib[(r + 8 * g) * 32 + n] = (_Float16)v;
        }
    }
    asm volatile("s_wait_dscnt 0x0" ::: "memory");
    // layer 4: [16x8] @ [8x4] (K,N padded)
    c = zero;
    c = wmma_f16(lds_load_A(ib, 32, 0, lane), lds_load_B_swz(w4f, lane), c);
    if (n < 4) {
        float bv = bs[40 + n];
#pragma unroll
        for (int r = 0; r < 8; ++r) cb[(r + 8 * g) * 4 + n] = c[r] + bv;
    }
    asm volatile("s_wait_dscnt 0x0" ::: "memory");
    if (lane < 16) {
        long eidx = e0 + lane;
        if (eidx < E) {
            float a0 = cb[lane * 4 + 0], a1 = cb[lane * 4 + 1];
            float a2 = cb[lane * 4 + 2], a3 = cb[lane * 4 + 3];
            float* er = e_raw + eidx * 4;
            er[0] = a0; er[1] = a1; er[2] = a2; er[3] = a3;
            float mx = fmaxf(fmaxf(a0, a1), fmaxf(a2, a3));
            float l = mx + logf(expf(a0 - mx) + expf(a1 - mx) + expf(a2 - mx) + expf(a3 - mx));
            float* oo = out_e + eidx * 4;
            oo[0] = a0 - l; oo[1] = a1 - l; oo[2] = a2 - l; oo[3] = a3 - l;
        }
    }
}

// ---------- GATConv 2 ----------
__global__ void node2_kernel(const float* x1, const float* W2, const float* as2,
                             const float* ad2, float* h2, float* hs2, float* hd2, int N) {
    int i = blockIdx.x * blockDim.x + threadIdx.x;
    if (i >= N) return;
    const float* xi = x1 + (size_t)i * 16;
    float h0 = 0.f, h1v = 0.f;
#pragma unroll
    for (int k = 0; k < 16; ++k) {
        float xv = xi[k];
        h0 += xv * W2[k * 2 + 0];
        h1v += xv * W2[k * 2 + 1];
    }
    h2[(size_t)i * 2 + 0] = h0;
    h2[(size_t)i * 2 + 1] = h1v;
    hs2[i] = h0 * as2[0] + h1v * as2[1];
    hd2[i] = h0 * ad2[0] + h1v * ad2[1];
}
__global__ void alpha2_kernel(const int* ei, const float* er, const float* hs2,
                              const float* hd2, const float* scal, float* alpha,
                              unsigned* menc, int E, int N) {
    int idx = blockIdx.x * blockDim.x + threadIdx.x;
    if (idx >= E + N) return;
    float v0 = scal[10], v1 = scal[11], v2 = scal[12], v3 = scal[13];
    int s, d; float et;
    if (idx < E) {
        s = ei[idx]; d = ei[(size_t)E + idx];
        const float* e_ = er + (size_t)idx * 4;
        et = e_[0] * v0 + e_[1] * v1 + e_[2] * v2 + e_[3] * v3;
    } else {
        int i = idx - E; s = i; d = i;
        float invE = 1.f / (float)E;
        et = (scal[3] * v0 + scal[4] * v1 + scal[5] * v2 + scal[6] * v3) * invE;
    }
    float a = leaky02(hs2[s] + hd2[d] + et);
    alpha[idx] = a;
    atomicMax(&menc[d], floatOrd(a));
}
__global__ void agg2_kernel(const int* ei, const float* p, const float* z,
                            const float* h2, float* x2, int E, int N) {
    int idx = blockIdx.x * blockDim.x + threadIdx.x;
    if (idx >= E + N) return;
    int s, d;
    if (idx < E) { s = ei[idx]; d = ei[(size_t)E + idx]; }
    else { s = idx - E; d = s; }
    float w = p[idx] / z[d];
    atomicAdd(&x2[(size_t)d * 2 + 0], w * h2[(size_t)s * 2 + 0]);
    atomicAdd(&x2[(size_t)d * 2 + 1], w * h2[(size_t)s * 2 + 1]);
}
__global__ void out_x2_kernel(const float* x2, float* out, int N) {
    int i = blockIdx.x * blockDim.x + threadIdx.x;
    if (i >= N) return;
    float a = x2[(size_t)i * 2 + 0], b = x2[(size_t)i * 2 + 1];
    float mx = fmaxf(a, b);
    float l = mx + logf(expf(a - mx) + expf(b - mx));
    out[(size_t)i * 2 + 0] = a - l;
    out[(size_t)i * 2 + 1] = b - l;
}

// ---------- launch ----------
extern "C" void kernel_launch(void* const* d_in, const int* in_sizes, int n_in,
                              void* d_out, int out_size, void* d_ws, size_t ws_size,
                              hipStream_t stream) {
    const float* x   = (const float*)d_in[0];
    const int*   ei  = (const int*)d_in[1];
    const float* ea  = (const float*)d_in[2];
    const float* W1  = (const float*)d_in[3];
    const float* as1 = (const float*)d_in[4];
    const float* ad1 = (const float*)d_in[5];
    const float* We1 = (const float*)d_in[6];
    const float* ae1 = (const float*)d_in[7];
    const float* b1  = (const float*)d_in[8];
    const float* Wm1 = (const float*)d_in[9];
    const float* bm1 = (const float*)d_in[10];
    const float* Wm2 = (const float*)d_in[11];
    const float* bm2 = (const float*)d_in[12];
    const float* Wm3 = (const float*)d_in[13];
    const float* bm3 = (const float*)d_in[14];
    const float* Wm4 = (const float*)d_in[15];
    const float* bm4 = (const float*)d_in[16];
    const float* W2  = (const float*)d_in[17];
    const float* as2 = (const float*)d_in[18];
    const float* ad2 = (const float*)d_in[19];
    const float* We2 = (const float*)d_in[20];
    const float* ae2 = (const float*)d_in[21];
    const float* b2  = (const float*)d_in[22];

    const int N = in_sizes[0] / 4;
    const int E = in_sizes[2] / 3;

    float* W = (float*)d_ws;
    size_t oH1 = 0;
    size_t oHS1 = oH1 + (size_t)N * 16;
    size_t oHD1 = oHS1 + (size_t)N;
    size_t oAL  = oHD1 + (size_t)N;
    size_t oM   = oAL + (size_t)(E + N);
    size_t oZ   = oM + (size_t)N;
    size_t oX1  = oZ + (size_t)N;
    size_t oER  = oX1 + (size_t)N * 16;
    size_t oH2  = oER + (size_t)E * 4;
    size_t oHS2 = oH2 + (size_t)N * 2;
    size_t oHD2 = oHS2 + (size_t)N;
    size_t oX2  = oHD2 + (size_t)N;
    size_t oSC  = oX2 + (size_t)N * 2;

    float*    h1    = W + oH1;
    float*    hs1   = W + oHS1;
    float*    hd1   = W + oHD1;
    float*    alpha = W + oAL;
    unsigned* menc  = (unsigned*)(W + oM);
    float*    z     = W + oZ;
    float*    x1    = W + oX1;
    float*    er    = W + oER;
    float*    h2    = W + oH2;
    float*    hs2   = W + oHS2;
    float*    hd2   = W + oHD2;
    float*    x2    = W + oX2;
    float*    scal  = W + oSC;

    float* outX = (float*)d_out;
    float* outE = outX + (size_t)N * 2;

    const int T = 256;
    int gN = (N + T - 1) / T;
    int gEN = (E + N + T - 1) / T;

    init1_kernel<<<gN, T, 0, stream>>>(menc, z, x1, x2, b1, b2, scal, N);
    precompute_kernel<<<1, 32, 0, stream>>>(We1, ae1, We2, ae2, scal);
    mean_ea_kernel<<<512, T, 0, stream>>>(ea, scal, E);
    node1_kernel<<<gN, T, 0, stream>>>(x, W1, as1, ad1, h1, hs1, hd1, N);
    alpha1_kernel<<<gEN, T, 0, stream>>>(ei, ea, hs1, hd1, scal, alpha, menc, E, N);
    exp_kernel<<<gEN, T, 0, stream>>>(ei, alpha, menc, z, E, N);
    agg1_kernel<<<gEN, T, 0, stream>>>(ei, alpha, z, h1, x1, E, N);

    long tiles = ((long)E + 15) / 16;
    int gMLP = (int)((tiles + 7) / 8);
    edge_mlp_kernel<<<gMLP, T, 0, stream>>>(ei, ea, x1, Wm1, bm1, Wm2, bm2,
                                            Wm3, bm3, Wm4, bm4, er, outE, E);

    mean_e_kernel<<<512, T, 0, stream>>>(er, scal, E);
    init2_kernel<<<gN, T, 0, stream>>>(menc, z, N);
    node2_kernel<<<gN, T, 0, stream>>>(x1, W2, as2, ad2, h2, hs2, hd2, N);
    alpha2_kernel<<<gEN, T, 0, stream>>>(ei, er, hs2, hd2, scal, alpha, menc, E, N);
    exp_kernel<<<gEN, T, 0, stream>>>(ei, alpha, menc, z, E, N);
    agg2_kernel<<<gEN, T, 0, stream>>>(ei, alpha, z, h2, x2, E, N);
    out_x2_kernel<<<gN, T, 0, stream>>>(x2, outX, N);
}
